// SwitchTransformersSparseMLP_25726854103354
// MI455X (gfx1250) — matrix operand order repaired and checked by reading, not compiled, not run on hardware
//
#include <hip/hip_runtime.h>
#include <hip/hip_bf16.h>

// ---------------------------------------------------------------------------
// Switch-Transformer sparse MLP (top-1 MoE) for MI455X / gfx1250.
// bf16 WMMA (v_wmma_f32_16x16x32_bf16) for the two expert GEMMs, f32 accum.
// Weights converted to bf16 (134 MB -> fits the 192 MB L2). GEMM tiles are
// staged into LDS with gfx1250 async copies (global_load_async_to_lds_b128,
// ASYNCcnt double-buffered) and read back with conflict-free ds_load_b128.
// B=4, S=2048, D=1024, F=4096, E=8, C=320.
// ---------------------------------------------------------------------------

#define BB 4
#define SS 2048
#define DD 1024
#define FF 4096
#define EE 8
#define CC 320

#define KSTEP 32
#define PAD   40      // padded K-stride (elements) in LDS: 80B -> conflict-free

typedef __bf16 bf16;
typedef __attribute__((ext_vector_type(16))) __bf16 v16bf;
typedef __attribute__((ext_vector_type(8)))  __bf16 v8bf;
typedef __attribute__((ext_vector_type(8)))  float  v8f;

// ---------------------------------------------------------------------------
// Router: one wave per token. 8 dot products over D, softmax/argmax.
// ---------------------------------------------------------------------------
__global__ void router_kernel(const float* __restrict__ x,
                              const float* __restrict__ rw,
                              float* __restrict__ logits_out,
                              int*   __restrict__ etok,
                              float* __restrict__ maxp) {
    int wave = (int)((blockIdx.x * blockDim.x + threadIdx.x) >> 5);
    int lane = threadIdx.x & 31;
    if (wave >= BB * SS) return;
    const float* xr = x + (size_t)wave * DD;

    float acc[EE];
#pragma unroll
    for (int e = 0; e < EE; ++e) acc[e] = 0.0f;

    for (int d = lane; d < DD; d += 32) {
        float xv = xr[d];
#pragma unroll
        for (int e = 0; e < EE; ++e) acc[e] += xv * rw[e * DD + d];
    }
#pragma unroll
    for (int off = 16; off > 0; off >>= 1) {
#pragma unroll
        for (int e = 0; e < EE; ++e) acc[e] += __shfl_down(acc[e], off, 32);
    }
    if (lane == 0) {
        float m = acc[0]; int am = 0;
#pragma unroll
        for (int e = 1; e < EE; ++e) { if (acc[e] > m) { m = acc[e]; am = e; } }
        float s = 0.0f;
#pragma unroll
        for (int e = 0; e < EE; ++e) s += __expf(acc[e] - m);
        maxp[wave] = 1.0f / s;       // softmax max prob = exp(0)/sum
        etok[wave] = am;
#pragma unroll
        for (int e = 0; e < EE; ++e) logits_out[(size_t)wave * EE + e] = acc[e];
    }
}

// ---------------------------------------------------------------------------
// Deterministic capacity assignment: one wave per batch, ballot prefix-sum.
// ---------------------------------------------------------------------------
__global__ void assign_kernel(const int* __restrict__ etok,
                              int* __restrict__ tok_slot,
                              int* __restrict__ slot_map,
                              int* __restrict__ expert_index_out) {
    int b = blockIdx.x;
    int lane = threadIdx.x;   // blockDim == 32

    for (int i = lane; i < EE * CC; i += 32) {
        int e = i / CC, c = i % CC;
        slot_map[(e * BB + b) * CC + c] = -1;
    }

    int cnt[EE];
#pragma unroll
    for (int e = 0; e < EE; ++e) cnt[e] = 0;

    for (int s0 = 0; s0 < SS; s0 += 32) {
        int t = b * SS + s0 + lane;
        int et = etok[t];
        int mypos = 0;
#pragma unroll
        for (int e = 0; e < EE; ++e) {
            unsigned mask  = (unsigned)__ballot(et == e);
            unsigned below = mask & ((1u << lane) - 1u);
            if (et == e) mypos = cnt[e] + __popc(below);
            cnt[e] += __popc(mask);
        }
        bool keep = (mypos < CC);
        if (keep) {
            slot_map[(et * BB + b) * CC + mypos] = t;
            tok_slot[t] = b * CC + mypos;
        } else {
            tok_slot[t] = -1;
        }
        expert_index_out[t] = keep ? et : 0;
    }
}

// ---------------------------------------------------------------------------
// f32 -> bf16 convert (weights), float4 vectorized.
// ---------------------------------------------------------------------------
__global__ void cvt_bf16_kernel(const float* __restrict__ in,
                                bf16* __restrict__ out, size_t n4) {
    size_t i = (size_t)blockIdx.x * blockDim.x + threadIdx.x;
    if (i >= n4) return;
    float4 v = *(const float4*)(in + i * 4);
    bf16* o = out + i * 4;
    o[0] = (bf16)v.x; o[1] = (bf16)v.y; o[2] = (bf16)v.z; o[3] = (bf16)v.w;
}

// ---------------------------------------------------------------------------
// Dispatch: gather kept tokens into bf16 expert buffers [E][B*C][D].
// ---------------------------------------------------------------------------
__global__ void dispatch_kernel(const float* __restrict__ x,
                                const int* __restrict__ slot_map,
                                bf16* __restrict__ xd) {
    int row  = (int)((blockIdx.x * blockDim.x + threadIdx.x) >> 5);
    int lane = threadIdx.x & 31;
    if (row >= EE * BB * CC) return;
    int t = slot_map[row];
    bf16* dst = xd + (size_t)row * DD;
    if (t >= 0) {
        const float* src = x + (size_t)t * DD;
        for (int d = lane; d < DD; d += 32) dst[d] = (bf16)src[d];
    } else {
        for (int d = lane; d < DD; d += 32) dst[d] = (bf16)0.0f;
    }
}

// ---------------------------------------------------------------------------
// gfx1250 async copy: 16B per lane, global -> LDS, tracked with ASYNCcnt.
// vdst = 32-bit LDS byte offset VGPR; vaddr = 64-bit global address.
// ---------------------------------------------------------------------------
__device__ __forceinline__ void async_b128(unsigned ldsoff, const void* gaddr) {
    asm volatile("global_load_async_to_lds_b128 %0, %1, off"
                 :: "v"(ldsoff), "v"(gaddr) : "memory");
}
__device__ __forceinline__ unsigned lds_off(const void* p) {
    return (unsigned)(uintptr_t)p;   // low 32 bits of generic ptr = LDS offset
}

// Stage one K-slab (KSTEP=32): A tile 128x32 and B tile 256x32 into LDS.
// 6 async b128 per thread-slot -> 6 per wave outstanding per stage.
__device__ __forceinline__ void stage_copy(const bf16* __restrict__ Ablk, int lda,
                                           const bf16* __restrict__ Bblk, int ldb,
                                           bf16* sA, bf16* sB, int tid) {
    // B: 256 columns, each 32 K-contiguous elements (64B = 4 x b128)
    {
        const bf16* g = Bblk + (size_t)tid * ldb;
        unsigned l = lds_off(sB + tid * PAD);
        async_b128(l,      g);
        async_b128(l + 16, g + 8);
        async_b128(l + 32, g + 16);
        async_b128(l + 48, g + 24);
    }
    // A: 128 rows x 64B; 512 b128 slots -> 2 per thread
    {
        int r0 = tid >> 2, q = tid & 3;
        async_b128(lds_off(sA + r0 * PAD + q * 8),
                   Ablk + (size_t)r0 * lda + q * 8);
        async_b128(lds_off(sA + (r0 + 64) * PAD + q * 8),
                   Ablk + (size_t)(r0 + 64) * lda + q * 8);
    }
}

// ---------------------------------------------------------------------------
// WMMA fragment loads from LDS (layout per CDNA5 ISA §7.12.2, wave32).
// A 16x32: lane<16 row holds K {0..7,16..23}; lane>=16 {8..15,24..31}.
// B 32x16: lane<16 col holds K {0..7,16..23}; lane>=16 {8..15,24..31}. PAD=40 elem stride.
// ---------------------------------------------------------------------------
__device__ __forceinline__ v16bf lds_a_frag(const bf16* sA, int row, int half) {
    const bf16* p = sA + row * PAD + half * 8;
    v8bf c0 = *(const v8bf*)(p);
    v8bf c1 = *(const v8bf*)(p + 16);
    v16bf r;
#pragma unroll
    for (int i = 0; i < 8; ++i) { r[i] = c0[i]; r[i + 8] = c1[i]; }
    return r;
}
__device__ __forceinline__ v16bf lds_b_frag(const bf16* sB, int col, int half) {
    const bf16* p = sB + col * PAD + half * 8;
    v8bf c0 = *(const v8bf*)(p);
    v8bf c1 = *(const v8bf*)(p + 16);
    v16bf r;
#pragma unroll
    for (int i = 0; i < 8; ++i) { r[i] = c0[i]; r[i + 8] = c1[i]; }
    return r;
}

// ---------------------------------------------------------------------------
// Batched expert GEMM: D = A[MxK] * B^T (B stored K-contiguous per column).
// Block 256 threads = 8 waves; block tile 128x256; wave tile 64x64
// (4x4 grid of 16x16x32 bf16 WMMA). Double-buffered async LDS staging.
// ---------------------------------------------------------------------------
template <bool RELU_BF16>
__global__ void __launch_bounds__(256)
wmma_gemm_kernel(const bf16* __restrict__ Aall, const bf16* __restrict__ Ball,
                 void* __restrict__ Call,
                 int K, int lda, int ldb, int ldc,
                 size_t strideA, size_t strideB, size_t strideC) {
    __shared__ __align__(16) bf16 sA[2][128 * PAD];   // 20 KB
    __shared__ __align__(16) bf16 sB[2][256 * PAD];   // 40 KB

    const int e = blockIdx.z;
    const bf16* A  = Aall + (size_t)e * strideA + (size_t)(blockIdx.x * 128) * lda;
    const bf16* Bw = Ball + (size_t)e * strideB + (size_t)(blockIdx.y * 256) * ldb;

    const int tid  = threadIdx.x;
    const int wave = tid >> 5;
    const int lane = tid & 31;
    const int r    = lane & 15;
    const int half = lane >> 4;
    const int wm   = wave & 1;       // 2 waves along M
    const int wn   = wave >> 1;      // 4 waves along N
    const int m0   = blockIdx.x * 128 + wm * 64;
    const int n0   = blockIdx.y * 256 + wn * 64;

    v8f acc[4][4];
#pragma unroll
    for (int mi = 0; mi < 4; ++mi)
#pragma unroll
        for (int nj = 0; nj < 4; ++nj)
#pragma unroll
            for (int i = 0; i < 8; ++i) acc[mi][nj][i] = 0.0f;

    const int nk = K / KSTEP;
    stage_copy(A, lda, Bw, ldb, sA[0], sB[0], tid);            // prologue

    for (int i = 0; i < nk; ++i) {
        const int cur = i & 1;
        if (i + 1 < nk) {
            stage_copy(A + (size_t)(i + 1) * KSTEP, lda,
                       Bw + (size_t)(i + 1) * KSTEP, ldb,
                       sA[cur ^ 1], sB[cur ^ 1], tid);
            asm volatile("s_wait_asynccnt 0x6" ::: "memory");  // cur stage done
        } else {
            asm volatile("s_wait_asynccnt 0x0" ::: "memory");
        }
        __syncthreads();                                       // all waves' copies

        v16bf af[4], bf[4];
#pragma unroll
        for (int mi = 0; mi < 4; ++mi)
            af[mi] = lds_a_frag(sA[cur], wm * 64 + 16 * mi + r, half);
#pragma unroll
        for (int nj = 0; nj < 4; ++nj)
            bf[nj] = lds_b_frag(sB[cur], wn * 64 + 16 * nj + r, half);

#pragma unroll
        for (int mi = 0; mi < 4; ++mi)
#pragma unroll
            for (int nj = 0; nj < 4; ++nj)
                acc[mi][nj] = __builtin_amdgcn_wmma_f32_16x16x32_bf16(
                    /*neg_a=*/false, af[mi], /*neg_b=*/false, bf[nj],
                    /*c_mod=*/(short)0, acc[mi][nj],
                    /*reuse_a=*/false, /*reuse_b=*/false);

        __syncthreads();   // protect next-iter overwrite of the other buffer
    }

    // Epilogue. C layout: vgpr i -> row = tile_m + 8*half + i, col = tile_n + r.
    if (RELU_BF16) {
        bf16* O = (bf16*)Call + (size_t)e * strideC;
#pragma unroll
        for (int mi = 0; mi < 4; ++mi)
#pragma unroll
            for (int nj = 0; nj < 4; ++nj) {
                int row = m0 + 16 * mi + 8 * half;
                int col = n0 + 16 * nj + r;
#pragma unroll
                for (int i = 0; i < 8; ++i) {
                    float v = acc[mi][nj][i];
                    v = v > 0.0f ? v : 0.0f;
                    O[(size_t)(row + i) * ldc + col] = (bf16)v;
                }
            }
    } else {
        float* O = (float*)Call + (size_t)e * strideC;
#pragma unroll
        for (int mi = 0; mi < 4; ++mi)
#pragma unroll
            for (int nj = 0; nj < 4; ++nj) {
                int row = m0 + 16 * mi + 8 * half;
                int col = n0 + 16 * nj + r;
#pragma unroll
                for (int i = 0; i < 8; ++i)
                    O[(size_t)(row + i) * ldc + col] = acc[mi][nj][i];
            }
    }
}

// ---------------------------------------------------------------------------
// Combine: out[t] = max_prob[t] * (kept ? y[e,slot] : x[t]). One wave/token.
// ---------------------------------------------------------------------------
__global__ void combine_kernel(const float* __restrict__ x,
                               const float* __restrict__ y,
                               const int* __restrict__ etok,
                               const int* __restrict__ tok_slot,
                               const float* __restrict__ maxp,
                               float* __restrict__ out) {
    int t    = (int)((blockIdx.x * blockDim.x + threadIdx.x) >> 5);
    int lane = threadIdx.x & 31;
    if (t >= BB * SS) return;
    int slot = tok_slot[t];
    int et   = etok[t];
    float p  = maxp[t];
    const float* src = (slot >= 0)
        ? (y + ((size_t)et * (BB * CC) + slot) * DD)
        : (x + (size_t)t * DD);
    float* dst = out + (size_t)t * DD;
    for (int d = lane; d < DD; d += 32) dst[d] = p * src[d];
}

// ---------------------------------------------------------------------------
// Launcher
// ---------------------------------------------------------------------------
extern "C" void kernel_launch(void* const* d_in, const int* in_sizes, int n_in,
                              void* d_out, int out_size, void* d_ws, size_t ws_size,
                              hipStream_t stream) {
    const float* x   = (const float*)d_in[0];   // [B,S,D]
    const float* rw  = (const float*)d_in[1];   // [E,D]
    const float* wi  = (const float*)d_in[2];   // [E,F,D]
    const float* wo  = (const float*)d_in[3];   // [E,D,F]

    float* out_states = (float*)d_out;                                  // [B,S,D]
    float* out_logits = out_states + (size_t)BB * SS * DD;              // [B,S,E]
    int*   out_eidx   = (int*)(out_logits + (size_t)BB * SS * EE);      // [B,S]

    // workspace layout (all offsets 256B aligned)
    char* ws = (char*)d_ws;
    int*   etok     = (int*)  (ws + 0);                 //  32768 B
    float* maxp     = (float*)(ws + 32768);             //  32768 B
    int*   tok_slot = (int*)  (ws + 65536);             //  32768 B
    int*   slot_map = (int*)  (ws + 98304);             //  40960 B
    bf16*  wi_bf    = (bf16*) (ws + 139264);            //  67108864 B
    bf16*  wo_bf    = (bf16*) (ws + 139264 + 67108864);
    bf16*  xd       = (bf16*) (ws + 139264 + 2ull * 67108864);                  // 20971520 B
    bf16*  h1       = (bf16*) (ws + 139264 + 2ull * 67108864 + 20971520ull);    // 83886080 B
    float* yb       = (float*)(ws + 139264 + 2ull * 67108864 + 20971520ull + 83886080ull);

    // 1) router: 8192 tokens, 8 waves/block
    router_kernel<<<1024, 256, 0, stream>>>(x, rw, out_logits, etok, maxp);

    // 2) deterministic capacity assignment: one wave per batch
    assign_kernel<<<BB, 32, 0, stream>>>(etok, tok_slot, slot_map, out_eidx);

    // 3) weight conversion f32 -> bf16 (results fit in 192MB L2)
    {
        size_t n4 = (size_t)EE * FF * DD / 4;           // 8,388,608
        cvt_bf16_kernel<<<(unsigned)(n4 / 256), 256, 0, stream>>>(wi, wi_bf, n4);
        cvt_bf16_kernel<<<(unsigned)(n4 / 256), 256, 0, stream>>>(wo, wo_bf, n4);
    }

    // 4) dispatch tokens into bf16 expert buffers [E][B*C][D]
    dispatch_kernel<<<(EE * BB * CC) / 8, 256, 0, stream>>>(x, slot_map, xd);

    // 5) GEMM1: h1 = relu(xd @ wi^T)   M=1280 N=4096 K=1024
    {
        dim3 grid(1280 / 128, FF / 256, EE);
        wmma_gemm_kernel<true><<<grid, 256, 0, stream>>>(
            xd, wi_bf, (void*)h1,
            /*K=*/DD, /*lda=*/DD, /*ldb=*/DD, /*ldc=*/FF,
            (size_t)BB * CC * DD, (size_t)FF * DD, (size_t)BB * CC * FF);
    }

    // 6) GEMM2: y = h1 @ wo^T          M=1280 N=1024 K=4096
    {
        dim3 grid(1280 / 128, DD / 256, EE);
        wmma_gemm_kernel<false><<<grid, 256, 0, stream>>>(
            h1, wo_bf, (void*)yb,
            /*K=*/FF, /*lda=*/FF, /*ldb=*/FF, /*ldc=*/DD,
            (size_t)BB * CC * FF, (size_t)DD * FF, (size_t)BB * CC * DD);
    }

    // 7) combine + scale by max router prob
    combine_kernel<<<1024, 256, 0, stream>>>(x, yb, etok, tok_slot, maxp, out_states);

    (void)in_sizes; (void)n_in; (void)out_size; (void)ws_size;
}
